// MultiValGCN_11647951307438
// MI455X (gfx1250) — matrix operand order repaired and checked by reading, not compiled, not run on hardware
//
#include <hip/hip_runtime.h>

typedef __bf16 bf16;
typedef __attribute__((ext_vector_type(16))) __bf16 v16bf;
typedef __attribute__((ext_vector_type(8)))  __bf16 v8bf;
typedef __attribute__((ext_vector_type(8)))  float  v8f;
typedef __attribute__((ext_vector_type(4)))  unsigned int v4u;
typedef __attribute__((ext_vector_type(8)))  int  v8i;
typedef __attribute__((ext_vector_type(4)))  int  v4i;

static constexpr int NN   = 8192;
static constexpr int DIN  = 512;
static constexpr int DHID = 1024;
static constexpr int DOUT = 64;

#define WMMA_BF16(a, b, c) \
  __builtin_amdgcn_wmma_f32_16x16x32_bf16(false, (a), false, (b), (short)0, (c), false, false)

union AF { v16bf v; v8bf h[2]; };

// ---------------------------------------------------------------------------
// Fragment layouts (wave32, CDNA5 ISA 7.12.2); LDS rows padded to 40 bf16.
// ---------------------------------------------------------------------------
__device__ inline v16bf load_a_frag(const bf16* As, int row, int half) {
  AF a;
  a.h[0] = *(const v8bf*)(As + row * 40 + 8 * half);
  a.h[1] = *(const v8bf*)(As + row * 40 + 16 + 8 * half);
  return a.v;
}
__device__ inline v16bf load_b_frag(const bf16* Bs, int col, int half) {
  AF b;
  b.h[0] = *(const v8bf*)(Bs + col * 40 + 8 * half);
  b.h[1] = *(const v8bf*)(Bs + col * 40 + 16 + 8 * half);
  return b.v;
}

struct AReg { float4 f[4]; };
__device__ inline AReg fetch_a(const float* gp) {
  AReg r;
#pragma unroll
  for (int j = 0; j < 4; j++) r.f[j] = ((const float4*)gp)[j];
  return r;
}
__device__ inline void stage_a(const AReg& r, bf16* dst) {
  AF u;
#pragma unroll
  for (int j = 0; j < 4; j++) {
    u.v[4 * j + 0] = (bf16)r.f[j].x; u.v[4 * j + 1] = (bf16)r.f[j].y;
    u.v[4 * j + 2] = (bf16)r.f[j].z; u.v[4 * j + 3] = (bf16)r.f[j].w;
  }
  *(v8bf*)dst = u.h[0]; *(v8bf*)(dst + 8) = u.h[1];
}

template <int NB>
__device__ inline void mma_tile(const bf16 (*As)[40], const bf16 (*Bs)[40],
                                v8f (*acc)[NB],
                                int wrow, int wcol, int lrow, int lhalf) {
  v16bf a[2], b[NB];
#pragma unroll
  for (int mi = 0; mi < 2; mi++)
    a[mi] = load_a_frag(&As[0][0], wrow * 32 + mi * 16 + lrow, lhalf);
#pragma unroll
  for (int ni = 0; ni < NB; ni++)
    b[ni] = load_b_frag(&Bs[0][0], wcol * (NB * 16) + ni * 16 + lrow, lhalf);
#pragma unroll
  for (int mi = 0; mi < 2; mi++)
#pragma unroll
    for (int ni = 0; ni < NB; ni++)
      acc[mi][ni] = WMMA_BF16(a[mi], b[ni], acc[mi][ni]);
}

// ---------------------------------------------------------------------------
// TDM: DMA a 2D bf16 tile (tile_d0 contiguous elems x tile_d1 rows of stride
// `stride_elems`) from global to LDS.  pad_interval=3 (every 16 DWORDs=32
// bf16) + pad_amount=3 (4 DWORDs=8 bf16) reproduces the 40-elem LDS row pitch.
// Descriptor bitfields per CDNA5 ISA 8.3-8.5 (count=1, type=2, data_size=2B).
// 6-arg builtin form (clang-23/therock): extra zero int32x8 group + cpol.
// ---------------------------------------------------------------------------
__device__ inline void tdm_load_2d_bf16(const void* gp, unsigned lds_addr,
                                        int tile_d0, int tile_d1,
                                        int stride_elems) {
  const unsigned long long ga = (unsigned long long)gp;
  v4u g0 = { 1u,                                   // count=1, user mode
             lds_addr,                             // LDS byte address
             (unsigned)ga,                         // global_addr[31:0]
             (unsigned)(ga >> 32) | (2u << 30) };  // global_addr[56:32] | type=2
  v8i g1 = { (int)((1u << 16) | (1u << 20) | (3u << 22) | (3u << 25)),
             (int)(((unsigned)NN & 0xFFFFu) << 16),   // tensor_dim0 = NN (lo16)
             (int)(((unsigned)NN & 0xFFFFu) << 16),   // dim0 hi=0 | tensor_dim1 lo16
             (int)((unsigned)tile_d0 << 16),          // dim1 hi=0 | tile_dim0
             (int)(unsigned)tile_d1,                  // tile_dim1, tile_dim2=0
             stride_elems,                            // tensor_dim0_stride lo32
             0, 0 };
  v4i gz4 = {0, 0, 0, 0};
  v8i gz8 = {0, 0, 0, 0, 0, 0, 0, 0};
  __builtin_amdgcn_tensor_load_to_lds(g0, g1, gz4, gz4, gz8, 0);
}
__device__ inline unsigned lds_off(const void* p) {
  return (unsigned)(unsigned long long)p;  // low 32 bits of generic ptr = LDS offset
}

// ---------------------------------------------------------------------------
// Stage 1: XW[k] = x @ W1[k]  (8192 x 512 x 1024) -> column-major bf16
// ---------------------------------------------------------------------------
__global__ __launch_bounds__(256) void k_xw(const float* __restrict__ x,
                                            const float* __restrict__ W1,
                                            bf16* __restrict__ xw_cm) {
  const int n0 = blockIdx.x * 128;
  const int o0 = blockIdx.y * 128;
  const int ord = blockIdx.z;
  const float* W = W1 + (size_t)ord * DIN * DHID;
  bf16* out = xw_cm + (size_t)ord * DHID * NN;

  __shared__ __align__(16) bf16 As[128][40];
  __shared__ __align__(16) bf16 Bs[128][40];

  const int t = threadIdx.x, lane = t & 31;
  const int wave = t >> 5, wrow = wave >> 1, wcol = wave & 1;
  const int lrow = lane & 15, lhalf = lane >> 4;
  const int aRow = t >> 1, aOff = (t & 1) * 16;

  v8f acc[2][4];
#pragma unroll
  for (int i = 0; i < 2; i++)
#pragma unroll
    for (int j = 0; j < 4; j++) acc[i][j] = {};

  for (int i0 = 0; i0 < DIN; i0 += 32) {
    AReg ra = fetch_a(x + (size_t)(n0 + aRow) * DIN + i0 + aOff);
    stage_a(ra, &As[aRow][aOff]);
#pragma unroll
    for (int q = t; q < 1024; q += 256) {  // transpose-load W1 tile 32x128
      const int r = q >> 5;
      const int cq = (q & 31) * 4;
      float4 f = *(const float4*)(W + (size_t)(i0 + r) * DHID + o0 + cq);
      Bs[cq + 0][r] = (bf16)f.x; Bs[cq + 1][r] = (bf16)f.y;
      Bs[cq + 2][r] = (bf16)f.z; Bs[cq + 3][r] = (bf16)f.w;
    }
    __syncthreads();
    mma_tile<4>(As, Bs, acc, wrow, wcol, lrow, lhalf);
    __syncthreads();
  }
#pragma unroll
  for (int mi = 0; mi < 2; mi++)
#pragma unroll
    for (int ni = 0; ni < 4; ni++) {
      const int col  = o0 + wcol * 64 + ni * 16 + lrow;
      const int rowb = n0 + wrow * 32 + mi * 16 + 8 * lhalf;
      AF u;
#pragma unroll
      for (int r = 0; r < 8; r++) u.v[r] = (bf16)acc[mi][ni][r];
      *(v8bf*)(out + (size_t)col * NN + rowb) = u.h[0];
    }
}

// ---------------------------------------------------------------------------
// Stage 2: H = relu( sum_k adj[k] @ XW[k] + b1 )
// grid(o0=x fast, n0=y): adj row-panels read once (L2 shares across o0),
// XW L2-resident.  Ping-pong LDS; A staged via VGPRs (needs f32->bf16),
// B tiles DMA'd by the Tensor Data Mover (wave 0 issues, TENSORcnt waits).
// ---------------------------------------------------------------------------
__global__ __launch_bounds__(256) void k_h(const float* __restrict__ adj,
                                           const bf16* __restrict__ xw_cm,
                                           const float* __restrict__ b1,
                                           bf16* __restrict__ h_rm) {
  const int o0 = blockIdx.x * 128;
  const int n0 = blockIdx.y * 128;

  __shared__ __align__(16) unsigned char smem[4 * 10240];  // 40960 B
  bf16 (*As0)[40] = (bf16(*)[40])(smem);
  bf16 (*Bs0)[40] = (bf16(*)[40])(smem + 10240);
  bf16 (*As1)[40] = (bf16(*)[40])(smem + 20480);
  bf16 (*Bs1)[40] = (bf16(*)[40])(smem + 30720);
  bf16 (*Ct)[136] = (bf16(*)[136])smem;  // epilogue reuse (34816 <= 40960)

  const int t = threadIdx.x, lane = t & 31;
  const int wave = t >> 5, wrow = wave >> 1, wcol = wave & 1;
  const int lrow = lane & 15, lhalf = lane >> 4;
  const int aRow = t >> 1, aOff = (t & 1) * 16;

  const float* base_a = adj + (size_t)(n0 + aRow) * NN + aOff;
  const bf16* base_b = xw_cm + (size_t)o0 * NN;   // tile base (col o0, m=0)

  v8f acc[2][4];
#pragma unroll
  for (int i = 0; i < 2; i++)
#pragma unroll
    for (int j = 0; j < 4; j++) acc[i][j] = {};

  constexpr int KIT = 3 * (NN / 32);  // 768, even
  auto ap = [&](int kk) {
    return base_a + (size_t)(kk >> 8) * NN * NN + (size_t)(kk & 255) * 32;
  };
  auto bgp = [&](int kk) {
    return base_b + (size_t)(kk >> 8) * ((size_t)DHID * NN) +
           (size_t)(kk & 255) * 32;
  };

  if (wave == 0) tdm_load_2d_bf16(bgp(0), lds_off(&Bs0[0][0]), 32, 128, NN);
  AReg ra = fetch_a(ap(0));
  stage_a(ra, &As0[aRow][aOff]);
  if (wave == 0) __builtin_amdgcn_s_wait_tensorcnt(0);
  __syncthreads();

  for (int kk = 0; kk < KIT; kk += 2) {
    if (wave == 0)
      tdm_load_2d_bf16(bgp(kk + 1), lds_off(&Bs1[0][0]), 32, 128, NN);
    ra = fetch_a(ap(kk + 1));
    mma_tile<4>(As0, Bs0, acc, wrow, wcol, lrow, lhalf);
    stage_a(ra, &As1[aRow][aOff]);
    if (wave == 0) __builtin_amdgcn_s_wait_tensorcnt(0);
    __syncthreads();

    if (kk + 2 < KIT) {
      if (wave == 0)
        tdm_load_2d_bf16(bgp(kk + 2), lds_off(&Bs0[0][0]), 32, 128, NN);
      ra = fetch_a(ap(kk + 2));
    }
    mma_tile<4>(As1, Bs1, acc, wrow, wcol, lrow, lhalf);
    if (kk + 2 < KIT) stage_a(ra, &As0[aRow][aOff]);
    if (wave == 0) __builtin_amdgcn_s_wait_tensorcnt(0);
    __syncthreads();
  }

  // epilogue: bias + relu -> LDS transpose -> coalesced row-major bf16 stores
#pragma unroll
  for (int mi = 0; mi < 2; mi++)
#pragma unroll
    for (int ni = 0; ni < 4; ni++) {
      const int col = wcol * 64 + ni * 16 + lrow;
      const float bv = b1[o0 + col];
      const int rowb = wrow * 32 + mi * 16 + 8 * lhalf;
#pragma unroll
      for (int r = 0; r < 8; r++) {
        float v = acc[mi][ni][r] + bv;
        v = v > 0.f ? v : 0.f;
        Ct[rowb + r][col] = (bf16)v;
      }
    }
  __syncthreads();
  {
    const int row = t >> 1;
    const int cb  = (t & 1) * 64;
    bf16* gp = h_rm + (size_t)(n0 + row) * DHID + o0 + cb;
#pragma unroll
    for (int j = 0; j < 64; j += 8)
      *(uint4*)(gp + j) = *(const uint4*)(&Ct[row][cb + j]);
  }
}

// ---------------------------------------------------------------------------
// Stage 3: HW[k] = H @ W2[k]  (8192 x 1024 x 64) -> column-major bf16
// ---------------------------------------------------------------------------
__global__ __launch_bounds__(256) void k_hw(const bf16* __restrict__ h_rm,
                                            const float* __restrict__ W2,
                                            bf16* __restrict__ hw_cm) {
  const int n0 = blockIdx.x * 128;
  const int ord = blockIdx.z;
  const float* W = W2 + (size_t)ord * DHID * DOUT;
  bf16* out = hw_cm + (size_t)ord * DOUT * NN;

  __shared__ __align__(16) bf16 As[128][40];
  __shared__ __align__(16) bf16 Bs[64][40];

  const int t = threadIdx.x, lane = t & 31;
  const int wave = t >> 5, wrow = wave >> 1, wcol = wave & 1;
  const int lrow = lane & 15, lhalf = lane >> 4;
  const int aRow = t >> 1, aOff = (t & 1) * 16;

  v8f acc[2][2];
#pragma unroll
  for (int i = 0; i < 2; i++)
#pragma unroll
    for (int j = 0; j < 2; j++) acc[i][j] = {};

  for (int h0 = 0; h0 < DHID; h0 += 32) {
    {
      const bf16* gp = h_rm + (size_t)(n0 + aRow) * DHID + h0 + aOff;
      uint4 d0 = ((const uint4*)gp)[0], d1 = ((const uint4*)gp)[1];
      *(uint4*)&As[aRow][aOff] = d0; *(uint4*)&As[aRow][aOff + 8] = d1;
    }
#pragma unroll
    for (int q = t; q < 512; q += 256) {  // transpose-load W2 tile 32x64
      const int r = q >> 4;
      const int cq = (q & 15) * 4;
      float4 f = *(const float4*)(W + (size_t)(h0 + r) * DOUT + cq);
      Bs[cq + 0][r] = (bf16)f.x; Bs[cq + 1][r] = (bf16)f.y;
      Bs[cq + 2][r] = (bf16)f.z; Bs[cq + 3][r] = (bf16)f.w;
    }
    __syncthreads();
    mma_tile<2>(As, Bs, acc, wrow, wcol, lrow, lhalf);
    __syncthreads();
  }
#pragma unroll
  for (int mi = 0; mi < 2; mi++)
#pragma unroll
    for (int ni = 0; ni < 2; ni++) {
      const int col  = wcol * 32 + ni * 16 + lrow;
      const int rowb = n0 + wrow * 32 + mi * 16 + 8 * lhalf;
      AF u;
#pragma unroll
      for (int r = 0; r < 8; r++) u.v[r] = (bf16)acc[mi][ni][r];
      *(v8bf*)(out + (size_t)col * NN + rowb) = u.h[0];
    }
}

// ---------------------------------------------------------------------------
// Stage 4a: zero the f32 accumulation buffer (ws is poisoned by the harness)
// ---------------------------------------------------------------------------
__global__ __launch_bounds__(256) void k_zero(float* __restrict__ p) {
  const int i = blockIdx.x * 256 + threadIdx.x;
  ((float4*)p)[i] = float4{0.f, 0.f, 0.f, 0.f};
}

// ---------------------------------------------------------------------------
// Stage 4b: partial OUT += adj_chunk @ HW_chunk  (K split 6 ways -> 384 blocks
// to saturate HBM on the final adj stream).  TDM for B, ping-pong LDS.
// ---------------------------------------------------------------------------
__global__ __launch_bounds__(256) void k_outp(const float* __restrict__ adj,
                                              const bf16* __restrict__ hw_cm,
                                              float* __restrict__ outacc) {
  const int n0 = blockIdx.x * 128;
  const int chunk = blockIdx.y;          // 0..5 = (ord, half)
  const int ord = chunk >> 1;
  const int mbase = (chunk & 1) * (NN / 2);

  __shared__ __align__(16) unsigned char smem[2 * 10240 + 2 * 5120];  // 30720 B
  bf16 (*As0)[40] = (bf16(*)[40])(smem);
  bf16 (*Bs0)[40] = (bf16(*)[40])(smem + 10240);
  bf16 (*As1)[40] = (bf16(*)[40])(smem + 15360);
  bf16 (*Bs1)[40] = (bf16(*)[40])(smem + 25600);

  const int t = threadIdx.x, lane = t & 31;
  const int wave = t >> 5, wrow = wave >> 1, wcol = wave & 1;
  const int lrow = lane & 15, lhalf = lane >> 4;
  const int aRow = t >> 1, aOff = (t & 1) * 16;

  const float* base_a = adj + (size_t)ord * NN * NN + (size_t)(n0 + aRow) * NN +
                        mbase + aOff;
  const bf16* base_b = hw_cm + (size_t)ord * DOUT * NN + mbase;

  v8f acc[2][2];
#pragma unroll
  for (int i = 0; i < 2; i++)
#pragma unroll
    for (int j = 0; j < 2; j++) acc[i][j] = {};

  constexpr int KIT = (NN / 2) / 32;  // 128, even
  if (wave == 0) tdm_load_2d_bf16(base_b, lds_off(&Bs0[0][0]), 32, 64, NN);
  AReg ra = fetch_a(base_a);
  stage_a(ra, &As0[aRow][aOff]);
  if (wave == 0) __builtin_amdgcn_s_wait_tensorcnt(0);
  __syncthreads();

  for (int kk = 0; kk < KIT; kk += 2) {
    if (wave == 0)
      tdm_load_2d_bf16(base_b + (size_t)(kk + 1) * 32, lds_off(&Bs1[0][0]),
                       32, 64, NN);
    ra = fetch_a(base_a + (size_t)(kk + 1) * 32);
    mma_tile<2>(As0, Bs0, acc, wrow, wcol, lrow, lhalf);
    stage_a(ra, &As1[aRow][aOff]);
    if (wave == 0) __builtin_amdgcn_s_wait_tensorcnt(0);
    __syncthreads();

    if (kk + 2 < KIT) {
      if (wave == 0)
        tdm_load_2d_bf16(base_b + (size_t)(kk + 2) * 32, lds_off(&Bs0[0][0]),
                         32, 64, NN);
      ra = fetch_a(base_a + (size_t)(kk + 2) * 32);
    }
    mma_tile<2>(As1, Bs1, acc, wrow, wcol, lrow, lhalf);
    if (kk + 2 < KIT) stage_a(ra, &As0[aRow][aOff]);
    if (wave == 0) __builtin_amdgcn_s_wait_tensorcnt(0);
    __syncthreads();
  }

#pragma unroll
  for (int mi = 0; mi < 2; mi++)
#pragma unroll
    for (int ni = 0; ni < 2; ni++) {
      const int col = wcol * 32 + ni * 16 + lrow;
      const int rowb = n0 + wrow * 32 + mi * 16 + 8 * lhalf;
#pragma unroll
      for (int r = 0; r < 8; r++)
        atomicAdd(outacc + (size_t)(rowb + r) * DOUT + col, acc[mi][ni][r]);
    }
}

// ---------------------------------------------------------------------------
// Stage 4c: bias + row-wise log-softmax, one thread per row
// ---------------------------------------------------------------------------
__global__ __launch_bounds__(256) void k_lsm(const float* __restrict__ outacc,
                                             const float* __restrict__ b2,
                                             float* __restrict__ out) {
  const int row = blockIdx.x * 256 + threadIdx.x;
  const float* rp = outacc + (size_t)row * DOUT;
  float v[DOUT];
  float m = -3.4e38f;
#pragma unroll
  for (int o = 0; o < DOUT; o++) {
    v[o] = rp[o] + b2[o];
    m = fmaxf(m, v[o]);
  }
  float s = 0.f;
#pragma unroll
  for (int o = 0; o < DOUT; o++) s += expf(v[o] - m);
  const float ls = m + logf(s);
  float* gp = out + (size_t)row * DOUT;
#pragma unroll
  for (int o = 0; o < DOUT; o++) gp[o] = v[o] - ls;
}

// ---------------------------------------------------------------------------
extern "C" void kernel_launch(void* const* d_in, const int* in_sizes, int n_in,
                              void* d_out, int out_size, void* d_ws, size_t ws_size,
                              hipStream_t stream) {
  const float* x   = (const float*)d_in[0];
  const float* adj = (const float*)d_in[1];   // [3, 8192, 8192]
  const float* W1  = (const float*)d_in[2];   // [3, 512, 1024]
  const float* b1  = (const float*)d_in[3];
  const float* W2  = (const float*)d_in[4];   // [3, 1024, 64]
  const float* b2  = (const float*)d_in[5];
  float* out = (float*)d_out;                 // [8192, 64]

  // ws: XW cm bf16 (48MB) | H rm bf16 (16MB) | HW cm bf16 (3MB) | outacc f32 (2MB)
  bf16* xw = (bf16*)d_ws;
  bf16* h  = xw + (size_t)3 * DHID * NN;
  bf16* hw = h + (size_t)NN * DHID;
  float* outacc = (float*)(hw + (size_t)3 * DOUT * NN);

  dim3 blk(256);
  k_xw  <<<dim3(NN / 128, DHID / 128, 3), blk, 0, stream>>>(x, W1, xw);
  k_h   <<<dim3(DHID / 128, NN / 128),    blk, 0, stream>>>(adj, xw, b1, h);
  k_hw  <<<dim3(NN / 128, 1, 3),          blk, 0, stream>>>(h, W2, hw);
  k_zero<<<dim3(NN * DOUT / 4 / 256),     blk, 0, stream>>>(outacc);
  k_outp<<<dim3(NN / 128, 6),             blk, 0, stream>>>(adj, hw, outacc);
  k_lsm <<<dim3(NN / 256),                blk, 0, stream>>>(outacc, b2, out);
}
